// SwinLayer_60163901882994
// MI455X (gfx1250) — matrix operand order repaired and checked by reading, not compiled, not run on hardware
//
#include <hip/hip_runtime.h>

// ---------------------------------------------------------------------------
// Types for CDNA5 WMMA (wave32): v_wmma_f32_16x16x32_bf16
// ---------------------------------------------------------------------------
typedef __bf16 bf16_t;
typedef __attribute__((ext_vector_type(16))) __bf16 v16bf;
typedef __attribute__((ext_vector_type(8)))  __bf16 v8bf;
typedef __attribute__((ext_vector_type(8)))  float  v8f;

#define DIM     512
#define HEADS   16
#define HD      32
#define HIDDEN  2048
#define NTOK    65536      // B*H*W = 4*128*128  (also B_*N)
#define NWIN    1024       // B_ windows
#define WN      64         // tokens per window
#define LDSP    40         // LDS row stride (elems) for 32-wide tiles (80B, 16B aligned)
#define LDSV    72         // LDS row stride for 64-wide tiles (144B, 16B aligned)

// ---------------------------------------------------------------------------
// Fragment loaders from LDS (layouts per CDNA5 ISA 7.12.2)
// A 16x32 bf16: lane L row=L&15, chunks K = (L/16)*8 and +16
// B 32x16 bf16: lane L col=L&15, chunks K = (L/16)*8 and +16
// ---------------------------------------------------------------------------
__device__ __forceinline__ v16bf frag_a(const bf16_t* s, int ld, int mbase, int kbase, int lane) {
    const bf16_t* p = s + (size_t)(mbase + (lane & 15)) * ld + kbase + ((lane >> 4) << 3);
    v8bf lo = *(const v8bf*)p;
    v8bf hi = *(const v8bf*)(p + 16);
    v16bf r;
#pragma unroll
    for (int i = 0; i < 8; ++i) { r[i] = lo[i]; r[i + 8] = hi[i]; }
    return r;
}
__device__ __forceinline__ v16bf frag_b(const bf16_t* s, int ld, int nbase, int kbase, int lane) {
    const bf16_t* p = s + (size_t)(nbase + (lane & 15)) * ld + kbase + ((lane >> 4) << 3);
    v8bf lo = *(const v8bf*)p;
    v8bf hi = *(const v8bf*)(p + 16);
    v16bf r;
#pragma unroll
    for (int i = 0; i < 8; ++i) { r[i] = lo[i]; r[i + 8] = hi[i]; }
    return r;
}
__device__ __forceinline__ v8f wmma_bf16(v16bf a, v16bf b, v8f c) {
    return __builtin_amdgcn_wmma_f32_16x16x32_bf16(false, a, false, b, (short)0, c, false, false);
}

// ---------------------------------------------------------------------------
// Weight conversion
// ---------------------------------------------------------------------------
__global__ void k_f32_to_bf16(const float* __restrict__ in, bf16_t* __restrict__ out, long n) {
    long i = (long)blockIdx.x * blockDim.x + threadIdx.x;
    if (i < n) out[i] = (bf16_t)in[i];
}
// fc1 rows interleaved: out row r <- src row (r odd ? HIDDEN + r/2 : r/2)
__global__ void k_fc1_interleave(const float* __restrict__ in, bf16_t* __restrict__ out) {
    long i = (long)blockIdx.x * blockDim.x + threadIdx.x;
    if (i < (long)2 * HIDDEN * DIM) {
        int k = (int)(i & (DIM - 1));
        int r = (int)(i >> 9);
        int src = (r & 1) ? (HIDDEN + (r >> 1)) : (r >> 1);
        out[i] = (bf16_t)in[(long)src * DIM + k];
    }
}

// ---------------------------------------------------------------------------
// LayerNorm (+ optional shifted-window gather), fp32 in -> bf16 out
// ---------------------------------------------------------------------------
__global__ __launch_bounds__(256) void k_ln(const float* __restrict__ src,
                                            bf16_t* __restrict__ dst,
                                            const float* __restrict__ gw,
                                            const float* __restrict__ bw,
                                            int windowed) {
    __shared__ float red[8];
    int m = blockIdx.x;
    long srow;
    if (windowed) {
        int w = m >> 6, n = m & 63;
        int bb = w >> 8, wi = w & 255;
        int wr = wi >> 4, wc = wi & 15;
        int r = n >> 3, c = n & 7;
        int p = wr * 8 + r, q = wc * 8 + c;
        int i = (p + 4) & 127, j = (q + 4) & 127;   // roll(-4): xs[p]=x[p+4]
        srow = ((long)bb * 128 + i) * 128 + j;
    } else {
        srow = m;
    }
    const float* xp = src + srow * DIM;
    int t = threadIdx.x, lane = t & 31, wv = t >> 5;
    float v0 = xp[t], v1 = xp[t + 256];
    float s = v0 + v1;
#pragma unroll
    for (int o = 16; o > 0; o >>= 1) s += __shfl_xor(s, o, 32);
    if (!lane) red[wv] = s;
    __syncthreads();
    float tot = 0.f;
#pragma unroll
    for (int i = 0; i < 8; ++i) tot += red[i];
    float mean = tot * (1.0f / 512.0f);
    __syncthreads();
    float d0 = v0 - mean, d1 = v1 - mean;
    s = d0 * d0 + d1 * d1;
#pragma unroll
    for (int o = 16; o > 0; o >>= 1) s += __shfl_xor(s, o, 32);
    if (!lane) red[wv] = s;
    __syncthreads();
    tot = 0.f;
#pragma unroll
    for (int i = 0; i < 8; ++i) tot += red[i];
    float rstd = rsqrtf(tot * (1.0f / 512.0f) + 1e-5f);
    dst[(long)m * DIM + t]       = (bf16_t)(d0 * rstd * gw[t] + bw[t]);
    dst[(long)m * DIM + t + 256] = (bf16_t)(d1 * rstd * gw[t + 256] + bw[t + 256]);
}

// ---------------------------------------------------------------------------
// Generic bf16 WMMA GEMM: out = A[M,K] @ W[N,K]^T + bias   (K compile-time)
// block tile 128x128, BK=32, 8 waves (4x2), wave tile 32x64 (2x4 WMMA)
// LDS double-buffered, one barrier per K-step; global loads overlap WMMA.
// ---------------------------------------------------------------------------
#define MODE_BF16    0
#define MODE_F32     1
#define MODE_F32ADD  2
#define MODE_SWIGLU  3

template <int MODE, int K>
__global__ __launch_bounds__(256) void k_gemm(const bf16_t* __restrict__ A,
                                              const bf16_t* __restrict__ Wt,
                                              const float*  __restrict__ bias,
                                              void* __restrict__ out,
                                              int ldo) {
    __shared__ __align__(16) bf16_t sA[2][128 * LDSP];
    __shared__ __align__(16) bf16_t sB[2][128 * LDSP];
    const int tid = threadIdx.x;
    const int lane = tid & 31;
    const int wave = tid >> 5;
    const int wm = wave >> 1;           // 0..3  -> M offset wm*32
    const int wn = wave & 1;            // 0..1  -> N offset wn*64
    const long mblk = (long)blockIdx.y * 128;
    const long nblk = (long)blockIdx.x * 128;

    __builtin_amdgcn_s_wait_tensorcnt(0);

    // staging layout: this thread owns rows r0 and r0+64, one 8-elem chunk
    const int r0   = tid >> 2;
    const int koff = (tid & 3) << 3;
    const bf16_t* gA = A  + (mblk + r0) * (long)K + koff;
    const bf16_t* gB = Wt + (nblk + r0) * (long)K + koff;
    constexpr long rstep = 64L * K;

    v8bf ra0 = *(const v8bf*)gA;
    v8bf ra1 = *(const v8bf*)(gA + rstep);
    v8bf rb0 = *(const v8bf*)gB;
    v8bf rb1 = *(const v8bf*)(gB + rstep);
    *(v8bf*)(sA[0] + r0 * LDSP + koff)        = ra0;
    *(v8bf*)(sA[0] + (r0 + 64) * LDSP + koff) = ra1;
    *(v8bf*)(sB[0] + r0 * LDSP + koff)        = rb0;
    *(v8bf*)(sB[0] + (r0 + 64) * LDSP + koff) = rb1;
    __syncthreads();

    v8f acc[2][4] = {};
    constexpr int nk = K >> 5;

#pragma unroll 2
    for (int kt = 0; kt < nk; ++kt) {
        const int cur = kt & 1, nxt = cur ^ 1;
        if (kt + 1 < nk) {
            const bf16_t* pa = gA + (long)(kt + 1) * 32;
            const bf16_t* pb = gB + (long)(kt + 1) * 32;
            ra0 = *(const v8bf*)pa;
            ra1 = *(const v8bf*)(pa + rstep);
            rb0 = *(const v8bf*)pb;
            rb1 = *(const v8bf*)(pb + rstep);
            if (kt + 2 < nk) {   // global_prefetch_b8 two tiles ahead
                __builtin_prefetch(gA + (long)(kt + 2) * 32, 0, 1);
                __builtin_prefetch(gA + (long)(kt + 2) * 32 + rstep, 0, 1);
            }
        }
        v16bf a0 = frag_a(sA[cur], LDSP, wm * 32,      0, lane);
        v16bf a1 = frag_a(sA[cur], LDSP, wm * 32 + 16, 0, lane);
#pragma unroll
        for (int j = 0; j < 4; ++j) {
            v16bf bj = frag_b(sB[cur], LDSP, wn * 64 + j * 16, 0, lane);
            acc[0][j] = wmma_bf16(a0, bj, acc[0][j]);
            acc[1][j] = wmma_bf16(a1, bj, acc[1][j]);
        }
        if (kt + 1 < nk) {
            *(v8bf*)(sA[nxt] + r0 * LDSP + koff)        = ra0;
            *(v8bf*)(sA[nxt] + (r0 + 64) * LDSP + koff) = ra1;
            *(v8bf*)(sB[nxt] + r0 * LDSP + koff)        = rb0;
            *(v8bf*)(sB[nxt] + (r0 + 64) * LDSP + koff) = rb1;
            __syncthreads();
        }
    }

#pragma unroll
    for (int i = 0; i < 2; ++i)
#pragma unroll
        for (int j = 0; j < 4; ++j)
#pragma unroll
            for (int e = 0; e < 8; ++e) {
                long row = mblk + wm * 32 + i * 16 + e + 8 * (lane >> 4);
                long col = nblk + wn * 64 + j * 16 + (lane & 15);
                float v = acc[i][j][e];
                if (MODE == MODE_BF16) {
                    ((bf16_t*)out)[row * ldo + col] = (bf16_t)(v + bias[col]);
                } else if (MODE == MODE_F32) {
                    ((float*)out)[row * ldo + col] = v + bias[col];
                } else if (MODE == MODE_F32ADD) {
                    float* po = (float*)out + row * ldo + col;
                    *po = *po + v + bias[col];
                } else { // MODE_SWIGLU: even col = a, odd col = gate (interleaved fc1)
                    float f = v + bias[((int)col & 1) * HIDDEN + ((int)col >> 1)];
                    float partner = __shfl_xor(f, 1, 32);
                    if (!((int)col & 1)) {
                        float g = partner;
                        float res = f * (g / (1.0f + __expf(-g)));
                        ((bf16_t*)out)[row * ldo + (col >> 1)] = (bf16_t)res;
                    }
                }
            }
}

// ---------------------------------------------------------------------------
// 2D RoPE on q,k of qkv buffer (in place, bf16); folds q scale (HD^-0.5)
// ---------------------------------------------------------------------------
__global__ __launch_bounds__(256) void k_rope(bf16_t* __restrict__ qkv) {
    int idx = blockIdx.x * 256 + threadIdx.x;
    int d   = idx & 7;
    int mat = (idx >> 3) & 1;
    int h   = (idx >> 4) & 15;
    int tok = idx >> 8;
    int n = tok & 63;
    int r = n >> 3, c = n & 7;
    bf16_t* p = qkv + (long)tok * (3 * DIM) + mat * DIM + h * HD;
    float x11 = (float)p[d];
    float x21 = (float)p[8 + d];
    float x12 = (float)p[16 + d];
    float x22 = (float)p[24 + d];
    float inv = exp2f(-(float)d * 1.66096404744f);  // 10000^(-d/8)
    float a1 = (float)r * inv, a2 = (float)c * inv;
    float s1 = sinf(a1), c1 = cosf(a1), s2 = sinf(a2), c2 = cosf(a2);
    float scale = (mat == 0) ? 0.17677669529f : 1.0f;  // 1/sqrt(32)
    p[d]      = (bf16_t)((x11 * c1 - x12 * s1) * scale);
    p[8 + d]  = (bf16_t)((x21 * c2 - x22 * s2) * scale);
    p[16 + d] = (bf16_t)((x12 * c1 + x11 * s1) * scale);
    p[24 + d] = (bf16_t)((x22 * c2 + x21 * s2) * scale);
}

// ---------------------------------------------------------------------------
// Windowed attention: one block per (window, head); 4 waves, wave = 16 rows.
// S = Q K^T (+mask) -> softmax -> P V;  out[token][h*32+d] bf16
// ---------------------------------------------------------------------------
__global__ __launch_bounds__(128) void k_attn(const bf16_t* __restrict__ qkv,
                                              const float*  __restrict__ mask,
                                              bf16_t* __restrict__ outp) {
    __shared__ __align__(16) bf16_t sQ[WN * LDSP];
    __shared__ __align__(16) bf16_t sK[WN * LDSP];
    __shared__ __align__(16) bf16_t sVt[HD * LDSV];
    __shared__ __align__(16) bf16_t sP[4][16 * LDSV];

    int bid = blockIdx.x;
    int h = bid & 15;
    int w = bid >> 4;
    int wi = w & 255;
    const bf16_t* qbase = qkv + (long)w * WN * (3 * DIM) + h * HD;
    int tid = threadIdx.x, lane = tid & 31, wv = tid >> 5;

#pragma unroll
    for (int it = 0; it < 2; ++it) {
        int cid = tid + 128 * it;
        int n = cid >> 2;
        int koff = (cid & 3) << 3;
        const bf16_t* row = qbase + (long)n * (3 * DIM);
        *(v8bf*)(sQ + n * LDSP + koff) = *(const v8bf*)(row + koff);
        *(v8bf*)(sK + n * LDSP + koff) = *(const v8bf*)(row + DIM + koff);
    }
    {   // V transposed: sVt[d][n]
        int n = tid >> 1;
        int dh = (tid & 1) << 4;
        const bf16_t* vp = qbase + (long)n * (3 * DIM) + 2 * DIM + dh;
#pragma unroll
        for (int d2 = 0; d2 < 16; ++d2) sVt[(dh + d2) * LDSV + n] = vp[d2];
    }
    __syncthreads();

    // scores: strip of 16 rows per wave
    v16bf aq = frag_a(sQ, LDSP, wv * 16, 0, lane);
    v8f sc[4];
#pragma unroll
    for (int mt = 0; mt < 4; ++mt) {
        v16bf bk = frag_b(sK, LDSP, mt * 16, 0, lane);
        v8f z = {};
        sc[mt] = wmma_bf16(aq, bk, z);
    }
    // mask + softmax (row held by a 16-lane half per VGPR element)
    const float* mrow = mask + (long)wi * WN * WN;
#pragma unroll
    for (int e = 0; e < 8; ++e) {
        int n = wv * 16 + e + 8 * (lane >> 4);
        float v[4];
        float mx = -1e30f;
#pragma unroll
        for (int mt = 0; mt < 4; ++mt) {
            int m = (lane & 15) + 16 * mt;
            v[mt] = sc[mt][e] + mrow[n * WN + m];
            mx = fmaxf(mx, v[mt]);
        }
#pragma unroll
        for (int o = 1; o < 16; o <<= 1) mx = fmaxf(mx, __shfl_xor(mx, o, 32));
        float sum = 0.f;
#pragma unroll
        for (int mt = 0; mt < 4; ++mt) { v[mt] = __expf(v[mt] - mx); sum += v[mt]; }
#pragma unroll
        for (int o = 1; o < 16; o <<= 1) sum += __shfl_xor(sum, o, 32);
        float invs = 1.0f / sum;
        int rloc = e + 8 * (lane >> 4);
#pragma unroll
        for (int mt = 0; mt < 4; ++mt)
            sP[wv][rloc * LDSV + (lane & 15) + 16 * mt] = (bf16_t)(v[mt] * invs);
    }
    __syncthreads();

    // P @ V : A = P (16x64, two K=32 chunks from sP), B from sVt
    v8f oacc[2] = {};
#pragma unroll
    for (int kc = 0; kc < 2; ++kc) {
        v16bf ap = frag_a(&sP[wv][0], LDSV, 0, kc * 32, lane);
#pragma unroll
        for (int dt = 0; dt < 2; ++dt) {
            v16bf bv = frag_b(sVt, LDSV, dt * 16, kc * 32, lane);
            oacc[dt] = wmma_bf16(ap, bv, oacc[dt]);
        }
    }
#pragma unroll
    for (int dt = 0; dt < 2; ++dt)
#pragma unroll
        for (int e = 0; e < 8; ++e) {
            int n = wv * 16 + e + 8 * (lane >> 4);
            int dcol = dt * 16 + (lane & 15);
            outp[((long)w * WN + n) * DIM + h * HD + dcol] = (bf16_t)oacc[dt][e];
        }
}

// ---------------------------------------------------------------------------
// un-shift / un-window + residual: out = x + scatter(proj)
// ---------------------------------------------------------------------------
__global__ __launch_bounds__(256) void k_unshift_res(const float* __restrict__ x,
                                                     const float* __restrict__ proj,
                                                     float* __restrict__ out) {
    long idx = (long)blockIdx.x * 256 + threadIdx.x;
    if (idx >= (long)NTOK * DIM) return;
    int c = (int)(idx & (DIM - 1));
    long pix = idx >> 9;
    int j = (int)(pix & 127);
    int i = (int)((pix >> 7) & 127);
    int bb = (int)(pix >> 14);
    int p = (i + 124) & 127;   // (i-4) mod 128
    int q = (j + 124) & 127;
    long row = (((long)bb * 16 + (p >> 3)) * 16 + (q >> 3)) * 64 + ((p & 7) * 8 + (q & 7));
    out[idx] = x[idx] + proj[row * DIM + c];
}

// ---------------------------------------------------------------------------
// Launch
// ---------------------------------------------------------------------------
extern "C" void kernel_launch(void* const* d_in, const int* in_sizes, int n_in,
                              void* d_out, int out_size, void* d_ws, size_t ws_size,
                              hipStream_t stream) {
    (void)in_sizes; (void)n_in; (void)out_size; (void)ws_size;
    const float* x      = (const float*)d_in[0];
    const float* mask   = (const float*)d_in[1];
    const float* n1g    = (const float*)d_in[2];
    const float* n1b    = (const float*)d_in[3];
    const float* qkv_w  = (const float*)d_in[4];
    const float* qkv_b  = (const float*)d_in[5];
    const float* proj_w = (const float*)d_in[6];
    const float* proj_b = (const float*)d_in[7];
    const float* n2g    = (const float*)d_in[8];
    const float* n2b    = (const float*)d_in[9];
    const float* fc1_w  = (const float*)d_in[10];
    const float* fc1_b  = (const float*)d_in[11];
    const float* fc2_w  = (const float*)d_in[12];
    const float* fc2_b  = (const float*)d_in[13];
    float* out = (float*)d_out;

    char* ws = (char*)d_ws;
    // weights (bf16)
    bf16_t* wqkv  = (bf16_t*)(ws + 0);                       // 1536*512
    bf16_t* wproj = (bf16_t*)(ws + 1572864);                 // 512*512
    bf16_t* wfc1  = (bf16_t*)(ws + 2097152);                 // 4096*512 interleaved
    bf16_t* wfc2  = (bf16_t*)(ws + 6291456);                 // 512*2048
    // activations
    bf16_t* bufA  = (bf16_t*)(ws + 8388608);                 // 64MiB: xw -> attnout -> y
    bf16_t* bqkv  = (bf16_t*)(ws + 8388608 + 67108864);      // 192MiB: qkv
    float*  tmpp  = (float*) (ws + 8388608 + 67108864 + 201326592);  // 128MiB proj out
    bf16_t* h2    = (bf16_t*)(ws + 8388608 + 67108864 + 201326592 + 134217728); // 256MiB

    // 1) weight conversion
    k_f32_to_bf16<<<(1536 * 512 + 255) / 256, 256, 0, stream>>>(qkv_w, wqkv, 1536L * 512);
    k_f32_to_bf16<<<(512 * 512 + 255) / 256, 256, 0, stream>>>(proj_w, wproj, 512L * 512);
    k_f32_to_bf16<<<(512 * 2048 + 255) / 256, 256, 0, stream>>>(fc2_w, wfc2, 512L * 2048);
    k_fc1_interleave<<<(4096 * 512 + 255) / 256, 256, 0, stream>>>(fc1_w, wfc1);

    // 2) LN1 + shift + window -> xw (bf16)
    k_ln<<<NTOK, 256, 0, stream>>>(x, bufA, n1g, n1b, 1);

    // 3) QKV GEMM: 65536x1536 = xw @ qkv_w^T + b (bf16 out)
    k_gemm<MODE_BF16, DIM><<<dim3(1536 / 128, NTOK / 128), 256, 0, stream>>>(
        bufA, wqkv, qkv_b, bqkv, 1536);

    // 4) RoPE on q,k (in place) + q scale
    k_rope<<<(NTOK * HEADS * 2 * 8) / 256, 256, 0, stream>>>(bqkv);

    // 5) attention per (window, head) -> attnout (bufA, bf16)
    k_attn<<<NWIN * HEADS, 128, 0, stream>>>(bqkv, mask, bufA);

    // 6) proj GEMM: 65536x512 fp32 out
    k_gemm<MODE_F32, DIM><<<dim3(DIM / 128, NTOK / 128), 256, 0, stream>>>(
        bufA, wproj, proj_b, tmpp, DIM);

    // 7) un-shift + residual -> d_out (= x_res)
    k_unshift_res<<<(NTOK * DIM) / 256, 256, 0, stream>>>(x, tmpp, out);

    // 8) LN2 -> y (bufA, bf16)
    k_ln<<<NTOK, 256, 0, stream>>>(out, bufA, n2g, n2b, 0);

    // 9) FC1 + SwiGLU (interleaved weights) -> h2 bf16 [65536 x 2048]
    k_gemm<MODE_SWIGLU, DIM><<<dim3(4096 / 128, NTOK / 128), 256, 0, stream>>>(
        bufA, wfc1, fc1_b, h2, HIDDEN);

    // 10) FC2, accumulate into d_out (residual)
    k_gemm<MODE_F32ADD, HIDDEN><<<dim3(DIM / 128, NTOK / 128), 256, 0, stream>>>(
        h2, wfc2, fc2_b, out, DIM);
}
